// _LightPointTransformerBlock_41729902248114
// MI455X (gfx1250) — matrix operand order repaired and checked by reading, not compiled, not run on hardware
//
#include <hip/hip_runtime.h>
#include <hip/hip_bf16.h>
#include <math.h>

// ---------------- problem constants ----------------
constexpr int cB = 2;
constexpr int cN = 32768;
constexpr int cK = 16;     // neighbors
constexpr int cD = 256;
constexpr int cA = 64;
constexpr int cV = 64;
constexpr int cF = 512;    // 2*D

typedef __attribute__((ext_vector_type(16))) _Float16 v16h;
typedef __attribute__((ext_vector_type(8)))  _Float16 v8h;
typedef __attribute__((ext_vector_type(8)))  float    v8f;

// ---------------- WMMA helpers (CDNA5 16x16x32 f16 -> f32) ----------------
__device__ __forceinline__ v8f wmma_f16(v16h a, v16h b, v8f c) {
  // 8 args: (neg_a, A, neg_b, B, c_mod, C, reuse_a, reuse_b)
  return __builtin_amdgcn_wmma_f32_16x16x32_f16(false, a, false, b, (short)0, c,
                                                false, false);
}

__device__ __forceinline__ v8f splat8(float v) {
  v8f r;
#pragma unroll
  for (int i = 0; i < 8; ++i) r[i] = v;
  return r;
}

// A fragment: 16x32 f16 tile from row-major buffer (ld halves, ld % 8 == 0,
// base 16B aligned). ISA 7.12.2: lanes 0-15 hold K 0..7 & 16..23 of row M=lane,
// lanes 16-31 hold K 8..15 & 24..31 of row M=lane-16.
__device__ __forceinline__ v16h load_a(const _Float16* buf, int ld, int kbase, int lane) {
  int row = lane & 15;
  int hb  = (lane >> 4) << 3;            // 0 or 8
  const _Float16* p = buf + row * ld + kbase + hb;
  v8h lo = *(const v8h*)(p);
  v8h hi = *(const v8h*)(p + 16);
  v16h a;
#pragma unroll
  for (int i = 0; i < 8; ++i) { a[i] = lo[i]; a[i + 8] = hi[i]; }
  return a;
}

// B fragment: 32x16 (K x N) tile. Weights are stored transposed as WT[N][Kd]
// so lane (col = lane&15) reads K values with two 16B loads.
// Lanes 0-15 hold K 0..7 & 16..23, lanes 16-31 hold K 8..15 & 24..31 (same as A).
__device__ __forceinline__ v16h load_b(const _Float16* wT, int Kd, int ntile,
                                       int kbase, int lane) {
  int n  = (ntile << 4) + (lane & 15);
  int kb = kbase + ((lane >> 4) << 3);
  const _Float16* p = wT + n * Kd + kb;
  v8h lo = *(const v8h*)(p);
  v8h hi = *(const v8h*)(p + 16);
  v16h b;
#pragma unroll
  for (int i = 0; i < 8; ++i) { b[i] = lo[i]; b[i + 8] = hi[i]; }
  return b;
}

__device__ __forceinline__ float wsum32(float v) {
  v += __shfl_xor(v, 1, 32);  v += __shfl_xor(v, 2, 32);
  v += __shfl_xor(v, 4, 32);  v += __shfl_xor(v, 8, 32);
  v += __shfl_xor(v, 16, 32);
  return v;
}

// ---------------- kernel 0: pack/transpose weights to f16 ----------------
__global__ __launch_bounds__(256) void k_prep(
    const float* Wq, const float* bq, const float* Wk, const float* bk,
    const float* Wv, const float* bv,
    const float* Wpa2, const float* Wpv2, const float* Ws1,
    const float* Wo, const float* Wf1, const float* Wf2,
    _Float16* WqkvT, float* bqkv,
    _Float16* Wpa2T, _Float16* Wpv2T, _Float16* Ws1T,
    _Float16* WoT, _Float16* Wf1T, _Float16* Wf2T) {
  int tid = blockIdx.x * 256 + threadIdx.x;     // grid covers 131072 threads
  if (tid < 192 * 256) {                        // WqkvT[c][d], Kd = 256
    int c = tid >> 8, d = tid & 255;
    float v = (c < 64) ? Wq[d * 64 + c]
            : (c < 128) ? Wk[d * 64 + (c - 64)]
                        : Wv[d * 64 + (c - 128)];
    WqkvT[c * 256 + d] = (_Float16)v;
  }
  if (tid < 192)
    bqkv[tid] = (tid < 64) ? bq[tid] : (tid < 128) ? bk[tid - 64] : bv[tid - 128];
  if (tid < 64 * 64) {                          // 64x64 mats: T[n][k] = W[k][n]
    int n = tid >> 6, k = tid & 63;
    Wpa2T[n * 64 + k] = (_Float16)Wpa2[k * 64 + n];
    Wpv2T[n * 64 + k] = (_Float16)Wpv2[k * 64 + n];
    Ws1T [n * 64 + k] = (_Float16)Ws1 [k * 64 + n];
  }
  if (tid < 256 * 64) {                         // Wo: 64(V) x 256(D) -> WoT[256][64]
    int n = tid >> 6, k = tid & 63;
    WoT[n * 64 + k] = (_Float16)Wo[k * 256 + n];
  }
  if (tid < 512 * 256) {                        // Wf1: 256 x 512 -> Wf1T[512][256]
    int n = tid >> 8, k = tid & 255;
    Wf1T[n * 256 + k] = (_Float16)Wf1[k * 512 + n];
  }
  if (tid < 256 * 512) {                        // Wf2: 512 x 256 -> Wf2T[256][512]
    int n = tid / 512, k = tid % 512;
    Wf2T[n * 512 + k] = (_Float16)Wf2[k * 256 + n];
  }
}

// ---------------- kernel 1: LayerNorm + fused QKV projection ----------------
// One block = 16 rows of (B*N). 8 waves: LN via shuffle reductions, then
// 12 16x16 output tiles (192 cols) with 8 WMMA k-steps each.
__global__ __launch_bounds__(256) void k_ln_qkv(
    const float* __restrict__ x, const float* __restrict__ g1,
    const float* __restrict__ b1,
    const _Float16* __restrict__ WqkvT, const float* __restrict__ bqkv,
    _Float16* __restrict__ qh, _Float16* __restrict__ kh,
    _Float16* __restrict__ vh) {
  __shared__ __align__(32) _Float16 hS[16][cD + 16];  // ld = 272
  int lane = threadIdx.x & 31, wid = threadIdx.x >> 5;
  int rt = blockIdx.x;

#pragma unroll
  for (int rr = 0; rr < 2; ++rr) {
    int rloc = wid * 2 + rr;
    long rg = (long)rt * 16 + rloc;
    const float* xr = x + rg * cD;
    float vals[8], s = 0.f, s2 = 0.f;
#pragma unroll
    for (int i = 0; i < 8; ++i) {
      float t = xr[lane * 8 + i];
      vals[i] = t; s += t; s2 += t * t;
    }
    s = wsum32(s); s2 = wsum32(s2);
    float mean = s * (1.f / cD);
    float var  = s2 * (1.f / cD) - mean * mean;
    float rstd = rsqrtf(var + 1e-5f);
#pragma unroll
    for (int i = 0; i < 8; ++i) {
      int d = lane * 8 + i;
      hS[rloc][d] = (_Float16)((vals[i] - mean) * rstd * g1[d] + b1[d]);
    }
  }
  __syncthreads();

  for (int ct = wid; ct < 12; ct += 8) {
    v8f acc = splat8(0.f);
#pragma unroll
    for (int kt = 0; kt < 8; ++kt)
      acc = wmma_f16(load_a(&hS[0][0], cD + 16, kt * 32, lane),
                     load_b(WqkvT, 256, ct, kt * 32, lane), acc);
    int col  = ct * 16 + (lane & 15);
    float bs = bqkv[col];
    int which = col >> 6, a64 = col & 63;
    _Float16* dst = (which == 0) ? qh : (which == 1) ? kh : vh;
#pragma unroll
    for (int r = 0; r < 8; ++r) {
      int m = r + ((lane >> 4) << 3);
      long rg = (long)rt * 16 + m;
      dst[rg * 64 + a64] = (_Float16)(acc[r] + bs);
    }
  }
}

// ---------------- kernel 2: attention (ra/rv MLP fused) ----------------
// Block = 128 threads (4 waves) handles 16 nodes of one batch. Each wave does
// 4 nodes sequentially; per node the 16 neighbors form the WMMA M dimension.
#define LDT 80
__global__ __launch_bounds__(128) void k_attn(
    const float* __restrict__ x, const int* __restrict__ knn_idx,
    const float* __restrict__ rel_pos,
    const _Float16* __restrict__ qh, const _Float16* __restrict__ kh,
    const _Float16* __restrict__ vh,
    const float* __restrict__ Wpa1, const float* __restrict__ bpa1,
    const float* __restrict__ bpa2,
    const float* __restrict__ Wpv1, const float* __restrict__ bpv1,
    const float* __restrict__ bpv2,
    const float* __restrict__ bs1, const float* __restrict__ Ws2,
    const float* __restrict__ bs2,
    const _Float16* __restrict__ Wpa2T, const _Float16* __restrict__ Wpv2T,
    const _Float16* __restrict__ Ws1T,
    const _Float16* __restrict__ WoT, const float* __restrict__ bo,
    float* __restrict__ out) {
  __shared__ __align__(16) _Float16 knS[4][16][64];
  __shared__ __align__(16) _Float16 vnS[4][16][64];
  __shared__ __align__(32) _Float16 tA[4][16][LDT];
  __shared__ __align__(32) _Float16 tT[4][16][LDT];
  __shared__ __align__(32) _Float16 ctxS[16][LDT];
  __shared__ float posS[4][16][4];
  __shared__ float scoreS[4][16];
  __shared__ float attnS[4][16];

  int lane = threadIdx.x & 31, wid = threadIdx.x >> 5;
  int bb = blockIdx.x / (cN / 16);
  int nt = blockIdx.x % (cN / 16);
  int n0 = nt * 16;

  for (int it = 0; it < 4; ++it) {
    int nloc = wid * 4 + it;
    int n = n0 + nloc;

    // --- stage neighbor data ---
    if (lane < 16) {
      const float* pp = rel_pos + ((long)n * cK + lane) * 3;
      posS[wid][lane][0] = pp[0];
      posS[wid][lane][1] = pp[1];
      posS[wid][lane][2] = pp[2];
    }
#pragma unroll
    for (int r = 0; r < 16; ++r) {
      int idx = knn_idx[n * cK + r];
      long base = ((long)bb * cN + idx) * 64;
      ((unsigned int*)&knS[wid][r][0])[lane] = ((const unsigned int*)(kh + base))[lane];
      ((unsigned int*)&vnS[wid][r][0])[lane] = ((const unsigned int*)(vh + base))[lane];
    }
    float qreg[4];
    {
      long qb = ((long)bb * cN + n) * 64;
#pragma unroll
      for (int ct = 0; ct < 4; ++ct)
        qreg[ct] = (float)qh[qb + ct * 16 + (lane & 15)];
    }
    __syncthreads();

    // --- ra layer 1 (3->64, VALU) into tA ---
    {
      int c0 = lane, c1 = lane + 32;
      float w00 = Wpa1[c0], w10 = Wpa1[64 + c0], w20 = Wpa1[128 + c0], bb0 = bpa1[c0];
      float w01 = Wpa1[c1], w11 = Wpa1[64 + c1], w21 = Wpa1[128 + c1], bb1 = bpa1[c1];
#pragma unroll
      for (int r = 0; r < 16; ++r) {
        float p0 = posS[wid][r][0], p1 = posS[wid][r][1], p2 = posS[wid][r][2];
        tA[wid][r][c0] = (_Float16)fmaxf(bb0 + p0 * w00 + p1 * w10 + p2 * w20, 0.f);
        tA[wid][r][c1] = (_Float16)fmaxf(bb1 + p0 * w01 + p1 * w11 + p2 * w21, 0.f);
      }
    }
    __syncthreads();

    // --- ra = relu1 @ Wpa2 (WMMA), fused tanh(q - kn + ra) -> tT ---
#pragma unroll
    for (int ct = 0; ct < 4; ++ct) {
      int cg = ct * 16 + (lane & 15);
      v8f acc = splat8(bpa2[cg]);
      acc = wmma_f16(load_a(&tA[wid][0][0], LDT, 0,  lane), load_b(Wpa2T, 64, ct, 0,  lane), acc);
      acc = wmma_f16(load_a(&tA[wid][0][0], LDT, 32, lane), load_b(Wpa2T, 64, ct, 32, lane), acc);
#pragma unroll
      for (int r = 0; r < 8; ++r) {
        int m = r + ((lane >> 4) << 3);
        float tv = tanhf(qreg[ct] - (float)knS[wid][m][cg] + acc[r]);
        tT[wid][m][cg] = (_Float16)tv;
      }
    }
    __syncthreads();

    // --- u = relu(t @ Ws1 + bs1) (WMMA); score = u @ Ws2 ---
    float part[8];
#pragma unroll
    for (int r = 0; r < 8; ++r) part[r] = 0.f;
#pragma unroll
    for (int ct = 0; ct < 4; ++ct) {
      int cg = ct * 16 + (lane & 15);
      v8f acc = splat8(bs1[cg]);
      acc = wmma_f16(load_a(&tT[wid][0][0], LDT, 0,  lane), load_b(Ws1T, 64, ct, 0,  lane), acc);
      acc = wmma_f16(load_a(&tT[wid][0][0], LDT, 32, lane), load_b(Ws1T, 64, ct, 32, lane), acc);
      float w2 = Ws2[cg];
#pragma unroll
      for (int r = 0; r < 8; ++r) part[r] += fmaxf(acc[r], 0.f) * w2;
    }
#pragma unroll
    for (int r = 0; r < 8; ++r) {           // reduce over the 16 lanes of each half
      float v = part[r];
      v += __shfl_xor(v, 1, 32); v += __shfl_xor(v, 2, 32);
      v += __shfl_xor(v, 4, 32); v += __shfl_xor(v, 8, 32);
      part[r] = v;
    }
    if ((lane & 15) == 0) {
      int hi = lane >> 4;
#pragma unroll
      for (int r = 0; r < 8; ++r) scoreS[wid][hi * 8 + r] = part[r] + bs2[0];
    }
    __syncthreads();

    // --- rv layer 1 into tA ---
    {
      int c0 = lane, c1 = lane + 32;
      float w00 = Wpv1[c0], w10 = Wpv1[64 + c0], w20 = Wpv1[128 + c0], bb0 = bpv1[c0];
      float w01 = Wpv1[c1], w11 = Wpv1[64 + c1], w21 = Wpv1[128 + c1], bb1 = bpv1[c1];
#pragma unroll
      for (int r = 0; r < 16; ++r) {
        float p0 = posS[wid][r][0], p1 = posS[wid][r][1], p2 = posS[wid][r][2];
        tA[wid][r][c0] = (_Float16)fmaxf(bb0 + p0 * w00 + p1 * w10 + p2 * w20, 0.f);
        tA[wid][r][c1] = (_Float16)fmaxf(bb1 + p0 * w01 + p1 * w11 + p2 * w21, 0.f);
      }
    }
    __syncthreads();

    // --- rv = relu1 @ Wpv2 (WMMA); vn += rv; softmax over K=16 ---
#pragma unroll
    for (int ct = 0; ct < 4; ++ct) {
      int cg = ct * 16 + (lane & 15);
      v8f acc = splat8(bpv2[cg]);
      acc = wmma_f16(load_a(&tA[wid][0][0], LDT, 0,  lane), load_b(Wpv2T, 64, ct, 0,  lane), acc);
      acc = wmma_f16(load_a(&tA[wid][0][0], LDT, 32, lane), load_b(Wpv2T, 64, ct, 32, lane), acc);
#pragma unroll
      for (int r = 0; r < 8; ++r) {
        int m = r + ((lane >> 4) << 3);
        vnS[wid][m][cg] = (_Float16)((float)vnS[wid][m][cg] + acc[r]);
      }
    }
    {
      float sc = scoreS[wid][lane & 15] * 0.125f;   // / sqrt(A=64)
      float mx = sc;
      mx = fmaxf(mx, __shfl_xor(mx, 1, 32)); mx = fmaxf(mx, __shfl_xor(mx, 2, 32));
      mx = fmaxf(mx, __shfl_xor(mx, 4, 32)); mx = fmaxf(mx, __shfl_xor(mx, 8, 32));
      float e = expf(sc - mx);
      float ss = e;
      ss += __shfl_xor(ss, 1, 32); ss += __shfl_xor(ss, 2, 32);
      ss += __shfl_xor(ss, 4, 32); ss += __shfl_xor(ss, 8, 32);
      if (lane < 16) attnS[wid][lane] = e / ss;
    }
    __syncthreads();

    // --- ctx = attn @ (vn + rv) ---
    {
      int c0 = lane, c1 = lane + 32;
      float a0 = 0.f, a1 = 0.f;
#pragma unroll
      for (int m = 0; m < 16; ++m) {
        float am = attnS[wid][m];
        a0 += am * (float)vnS[wid][m][c0];
        a1 += am * (float)vnS[wid][m][c1];
      }
      ctxS[nloc][c0] = (_Float16)a0;
      ctxS[nloc][c1] = (_Float16)a1;
    }
    __syncthreads();
  }

  // --- phase 2: out = x + ctx @ Wo + bo (16x64 @ 64x256, cooperative) ---
#pragma unroll
  for (int cti = 0; cti < 4; ++cti) {
    int ct = wid * 4 + cti;
    int cg = ct * 16 + (lane & 15);
    v8f acc = splat8(bo[cg]);
    acc = wmma_f16(load_a(&ctxS[0][0], LDT, 0,  lane), load_b(WoT, 64, ct, 0,  lane), acc);
    acc = wmma_f16(load_a(&ctxS[0][0], LDT, 32, lane), load_b(WoT, 64, ct, 32, lane), acc);
#pragma unroll
    for (int r = 0; r < 8; ++r) {
      int m = r + ((lane >> 4) << 3);
      long o = ((long)bb * cN + n0 + m) * cD + cg;
      out[o] = x[o] + acc[r];
    }
  }
}

// ---------------- kernel 3: LN2 + FFN (gelu) + residual, in-place on out ---
__global__ __launch_bounds__(256) void k_ffn(
    const float* __restrict__ g2, const float* __restrict__ b2,
    const _Float16* __restrict__ Wf1T, const float* __restrict__ bf1,
    const _Float16* __restrict__ Wf2T, const float* __restrict__ bf2,
    float* __restrict__ out) {
  __shared__ __align__(32) _Float16 hS[16][cD + 16];   // ld 272
  __shared__ __align__(32) _Float16 uS[16][cF + 16];   // ld 528
  int lane = threadIdx.x & 31, wid = threadIdx.x >> 5;
  int rt = blockIdx.x;

#pragma unroll
  for (int rr = 0; rr < 2; ++rr) {
    int rloc = wid * 2 + rr;
    long rg = (long)rt * 16 + rloc;
    const float* xr = out + rg * cD;
    float vals[8], s = 0.f, s2 = 0.f;
#pragma unroll
    for (int i = 0; i < 8; ++i) {
      float t = xr[lane * 8 + i];
      vals[i] = t; s += t; s2 += t * t;
    }
    s = wsum32(s); s2 = wsum32(s2);
    float mean = s * (1.f / cD);
    float var  = s2 * (1.f / cD) - mean * mean;
    float rstd = rsqrtf(var + 1e-5f);
#pragma unroll
    for (int i = 0; i < 8; ++i) {
      int d = lane * 8 + i;
      hS[rloc][d] = (_Float16)((vals[i] - mean) * rstd * g2[d] + b2[d]);
    }
  }
  __syncthreads();

  // u = gelu(h @ Wf1 + bf1): 32 col tiles over 8 waves
#pragma unroll
  for (int cti = 0; cti < 4; ++cti) {
    int ct = wid + cti * 8;
    int cg = ct * 16 + (lane & 15);
    v8f acc = splat8(bf1[cg]);
#pragma unroll
    for (int kt = 0; kt < 8; ++kt)
      acc = wmma_f16(load_a(&hS[0][0], cD + 16, kt * 32, lane),
                     load_b(Wf1T, 256, ct, kt * 32, lane), acc);
#pragma unroll
    for (int r = 0; r < 8; ++r) {
      int m = r + ((lane >> 4) << 3);
      float u = acc[r];
      uS[m][cg] = (_Float16)(0.5f * u * (1.f + erff(u * 0.70710678f)));
    }
  }
  __syncthreads();

  // out += u @ Wf2 + bf2: 16 col tiles over 8 waves
#pragma unroll
  for (int cti = 0; cti < 2; ++cti) {
    int ct = wid + cti * 8;
    int cg = ct * 16 + (lane & 15);
    v8f acc = splat8(bf2[cg]);
#pragma unroll
    for (int kt = 0; kt < 16; ++kt)
      acc = wmma_f16(load_a(&uS[0][0], cF + 16, kt * 32, lane),
                     load_b(Wf2T, 512, ct, kt * 32, lane), acc);
#pragma unroll
    for (int r = 0; r < 8; ++r) {
      int m = r + ((lane >> 4) << 3);
      long o = ((long)rt * 16 + m) * cD + cg;
      out[o] = out[o] + acc[r];
    }
  }
}

// ---------------- host launcher ----------------
extern "C" void kernel_launch(void* const* d_in, const int* in_sizes, int n_in,
                              void* d_out, int out_size, void* d_ws, size_t ws_size,
                              hipStream_t stream) {
  const float* x       = (const float*)d_in[0];
  const int*   knn_idx = (const int*)  d_in[1];
  const float* rel_pos = (const float*)d_in[2];
  const float* g1   = (const float*)d_in[3];
  const float* b1   = (const float*)d_in[4];
  const float* g2   = (const float*)d_in[5];
  const float* b2   = (const float*)d_in[6];
  const float* Wq   = (const float*)d_in[7];
  const float* bq   = (const float*)d_in[8];
  const float* Wk   = (const float*)d_in[9];
  const float* bk   = (const float*)d_in[10];
  const float* Wv   = (const float*)d_in[11];
  const float* bv   = (const float*)d_in[12];
  const float* Wo   = (const float*)d_in[13];
  const float* bo   = (const float*)d_in[14];
  const float* Wpa1 = (const float*)d_in[15];
  const float* bpa1 = (const float*)d_in[16];
  const float* Wpa2 = (const float*)d_in[17];
  const float* bpa2 = (const float*)d_in[18];
  const float* Wpv1 = (const float*)d_in[19];
  const float* bpv1 = (const float*)d_in[20];
  const float* Wpv2 = (const float*)d_in[21];
  const float* bpv2 = (const float*)d_in[22];
  const float* Ws1  = (const float*)d_in[23];
  const float* bs1  = (const float*)d_in[24];
  const float* Ws2  = (const float*)d_in[25];
  const float* bs2  = (const float*)d_in[26];
  const float* Wf1  = (const float*)d_in[27];
  const float* bf1  = (const float*)d_in[28];
  const float* Wf2  = (const float*)d_in[29];
  const float* bf2  = (const float*)d_in[30];
  (void)in_sizes; (void)n_in; (void)out_size; (void)ws_size;

  char* ws = (char*)d_ws;
  size_t off = 0;
  auto take = [&](size_t bytes) -> void* {
    void* p = ws + off;
    off += (bytes + 255) & ~(size_t)255;
    return p;
  };
  _Float16* qh    = (_Float16*)take(sizeof(_Float16) * (size_t)cB * cN * cA);
  _Float16* kh    = (_Float16*)take(sizeof(_Float16) * (size_t)cB * cN * cA);
  _Float16* vh    = (_Float16*)take(sizeof(_Float16) * (size_t)cB * cN * cV);
  _Float16* WqkvT = (_Float16*)take(sizeof(_Float16) * 192 * 256);
  float*    bqkv  = (float*)   take(sizeof(float) * 192);
  _Float16* Wpa2T = (_Float16*)take(sizeof(_Float16) * 64 * 64);
  _Float16* Wpv2T = (_Float16*)take(sizeof(_Float16) * 64 * 64);
  _Float16* Ws1T  = (_Float16*)take(sizeof(_Float16) * 64 * 64);
  _Float16* WoT   = (_Float16*)take(sizeof(_Float16) * 256 * 64);
  _Float16* Wf1T  = (_Float16*)take(sizeof(_Float16) * 512 * 256);
  _Float16* Wf2T  = (_Float16*)take(sizeof(_Float16) * 256 * 512);

  float* out = (float*)d_out;

  k_prep<<<512, 256, 0, stream>>>(Wq, bq, Wk, bk, Wv, bv, Wpa2, Wpv2, Ws1,
                                  Wo, Wf1, Wf2, WqkvT, bqkv, Wpa2T, Wpv2T,
                                  Ws1T, WoT, Wf1T, Wf2T);

  k_ln_qkv<<<(cB * cN) / 16, 256, 0, stream>>>(x, g1, b1, WqkvT, bqkv,
                                               qh, kh, vh);

  k_attn<<<cB * (cN / 16), 128, 0, stream>>>(
      x, knn_idx, rel_pos, qh, kh, vh,
      Wpa1, bpa1, bpa2, Wpv1, bpv1, bpv2,
      bs1, Ws2, bs2, Wpa2T, Wpv2T, Ws1T, WoT, bo, out);

  k_ffn<<<(cB * cN) / 16, 256, 0, stream>>>(g2, b2, Wf1T, bf1, Wf2T, bf2, out);
}
